// MQACrossAttention_2070174237276
// MI455X (gfx1250) — matrix operand order repaired and checked by reading, not compiled, not run on hardware
//
#include <hip/hip_runtime.h>
#include <hip/hip_bf16.h>

// ---------------------------------------------------------------------------
// MQA cross-attention for MI455X (gfx1250, wave32, WMMA).
//   B=4, LQ=LK=2048, D=1024, H=16, HD=64
// Pipeline:
//   1) gemm_proj_kernel (mode 0): Qh  = f16((x @ Wq + bq) * 1/8)   [b,h,lq,hd]
//   2) gemm_proj_kernel (mode 1): Kh  = f16(text @ Wkv_k + bk)     [b,h,lk,hd]
//                                 Vt  = f16(text @ Wkv_v + bv)^T   [b,h,hd,lk]
//   3) flash_attn_kernel: out = softmax(Qh Kh^T) Vt  (f32 accum, f32 out)
// All matrix math on v_wmma_f32_16x16x32_f16. Double-buffered LDS staging;
// flash K/V tiles staged via GLOBAL_LOAD_ASYNC_TO_LDS when available.
// ---------------------------------------------------------------------------

typedef __attribute__((ext_vector_type(16))) _Float16 v16h;
typedef __attribute__((ext_vector_type(8)))  _Float16 v8h;
typedef __attribute__((ext_vector_type(4)))  _Float16 v4h;
typedef __attribute__((ext_vector_type(8)))  float    v8f;
typedef __attribute__((ext_vector_type(4)))  float    v4f;
typedef __attribute__((ext_vector_type(4)))  int      v4i;

union HFrag {
    v16h v;
    struct { v8h lo; v8h hi; } p;
};

#define WMMA16(a, b, c) \
    __builtin_amdgcn_wmma_f32_16x16x32_f16(false, (a), false, (b), (short)0, (c), false, false)

#define NB  4
#define NH  16
#define SEQ 2048
#define DM  1024
#define HD  64

#if __has_builtin(__builtin_amdgcn_global_load_async_to_lds_b128) && \
    __has_builtin(__builtin_amdgcn_s_wait_asynccnt)
#define HAVE_ASYNC_LDS 1
#else
#define HAVE_ASYNC_LDS 0
#endif

#define AS1 __attribute__((address_space(1)))
#define AS3 __attribute__((address_space(3)))

// 16-byte global -> LDS copy; async DMA path when the toolchain exposes it.
__device__ __forceinline__ void cp16_g2l(const _Float16* g, _Float16* l)
{
#if HAVE_ASYNC_LDS
    __builtin_amdgcn_global_load_async_to_lds_b128((AS1 v4i*)g, (AS3 v4i*)l, 0, 0);
#else
    *(v8h*)l = *(const v8h*)g;
#endif
}

__device__ __forceinline__ void cp_fence_lds()
{
#if HAVE_ASYNC_LDS
    __builtin_amdgcn_s_wait_asynccnt(0);
#endif
}

// ---------------------------------------------------------------------------
// Projection GEMM: C[M,N] = A[M,K] @ W[K,N] + bias, output routed to f16
// head-split buffers. 256 threads = 8 waves; block tile 128(M) x 128(N),
// K-chunk 32, double-buffered LDS. Wave computes 64x32 via 4x2 WMMA tiles.
// ---------------------------------------------------------------------------
__global__ __launch_bounds__(256)
void gemm_proj_kernel(const float* __restrict__ A,
                      const float* __restrict__ W,
                      const float* __restrict__ bias,
                      _Float16* __restrict__ outQ,
                      _Float16* __restrict__ outK,
                      _Float16* __restrict__ outV,
                      int N, int Kdim, int mode)
{
    constexpr int LDA = 40;                    // padded stride (halves): conflict-free
    __shared__ _Float16 Alds[2][128 * LDA];    // A tile, row-major [m][k]
    __shared__ _Float16 Blds[2][128 * LDA];    // W tile, transposed [n][k]

    const int tid   = threadIdx.x;
    const int lane  = tid & 31;
    const int wid   = tid >> 5;                // 0..7
    const int waveM = wid & 1;                 // 2 waves along M  -> 64 rows each
    const int waveN = wid >> 1;                // 4 waves along N  -> 32 cols each
    const int g     = lane >> 4;               // 16-lane group
    const int l16   = lane & 15;

    const int mBase = blockIdx.y * 128;
    const int nBase = blockIdx.x * 128;

    // --- per-thread staging coordinates (fixed across K-steps) ---
    int aRow[4], aCv[4], wK[4], wCv[4];
    const float* aPtr[4];
    const float* wPtr[4];
#pragma unroll
    for (int i = 0; i < 4; ++i) {
        const int idx = tid + i * 256;         // 0..1023
        aRow[i] = idx >> 3;  aCv[i] = idx & 7;     // 128 rows x 8 vec4
        wK[i]   = idx >> 5;  wCv[i] = idx & 31;    // 32 rows x 32 vec4
        aPtr[i] = A + (size_t)(mBase + aRow[i]) * Kdim + aCv[i] * 4;
        wPtr[i] = W + (size_t)wK[i] * N + nBase + wCv[i] * 4;
    }

    v4f aReg[4], wReg[4];
    auto loadRegs = [&]() {                     // issue all 8 b128 loads, no waits
#pragma unroll
        for (int i = 0; i < 4; ++i) aReg[i] = *(const v4f*)aPtr[i];
#pragma unroll
        for (int i = 0; i < 4; ++i) wReg[i] = *(const v4f*)wPtr[i];
    };
    auto storeTile = [&](int buf) {             // convert + LDS store
#pragma unroll
        for (int i = 0; i < 4; ++i) {
            v4h h;
            h.x = (_Float16)aReg[i].x; h.y = (_Float16)aReg[i].y;
            h.z = (_Float16)aReg[i].z; h.w = (_Float16)aReg[i].w;
            *(v4h*)&Alds[buf][aRow[i] * LDA + aCv[i] * 4] = h;
        }
#pragma unroll
        for (int i = 0; i < 4; ++i) {           // transpose into [n][k]
            Blds[buf][(wCv[i] * 4 + 0) * LDA + wK[i]] = (_Float16)wReg[i].x;
            Blds[buf][(wCv[i] * 4 + 1) * LDA + wK[i]] = (_Float16)wReg[i].y;
            Blds[buf][(wCv[i] * 4 + 2) * LDA + wK[i]] = (_Float16)wReg[i].z;
            Blds[buf][(wCv[i] * 4 + 3) * LDA + wK[i]] = (_Float16)wReg[i].w;
        }
    };

    v8f acc[4][2];
#pragma unroll
    for (int mt = 0; mt < 4; ++mt)
#pragma unroll
        for (int nt = 0; nt < 2; ++nt)
            acc[mt][nt] = (v8f){0.f, 0.f, 0.f, 0.f, 0.f, 0.f, 0.f, 0.f};

    // prologue: stage K-step 0 into buffer 0
    loadRegs();
    storeTile(0);
    __syncthreads();

    const int nIter = Kdim >> 5;               // K / 32
    for (int it = 0; it < nIter; ++it) {
        const int  cur  = it & 1;
        const bool more = (it + 1) < nIter;
        if (more) {                            // issue next-tile loads early
#pragma unroll
            for (int i = 0; i < 4; ++i) { aPtr[i] += 32; wPtr[i] += (size_t)32 * N; }
            loadRegs();
        }

        // ---- fragments + WMMA on current buffer ----
        HFrag a[4];
#pragma unroll
        for (int mt = 0; mt < 4; ++mt) {
            const int row = waveM * 64 + mt * 16 + l16;
            a[mt].p.lo = *(const v8h*)&Alds[cur][row * LDA + g * 8];       // K = 8g+0..7
            a[mt].p.hi = *(const v8h*)&Alds[cur][row * LDA + 16 + g * 8];  // K = 16+8g+..
        }
        HFrag b[2];
#pragma unroll
        for (int nt = 0; nt < 2; ++nt) {
            const int n = waveN * 32 + nt * 16 + l16;
            b[nt].p.lo = *(const v8h*)&Blds[cur][n * LDA + g * 8];         // K = 8g+0..7
            b[nt].p.hi = *(const v8h*)&Blds[cur][n * LDA + 16 + g * 8];    // K = 16+8g+..
        }
#pragma unroll
        for (int mt = 0; mt < 4; ++mt)
#pragma unroll
            for (int nt = 0; nt < 2; ++nt)
                acc[mt][nt] = WMMA16(a[mt].v, b[nt].v, acc[mt][nt]);

        if (more) storeTile(cur ^ 1);          // waits loadcnt only here
        __syncthreads();
    }

    // ---- epilogue: bias + head-split routing, f16 store ----
#pragma unroll
    for (int mt = 0; mt < 4; ++mt) {
#pragma unroll
        for (int nt = 0; nt < 2; ++nt) {
            const int nc = nBase + waveN * 32 + nt * 16 + l16;
            const float bv = bias[nc];
#pragma unroll
            for (int r = 0; r < 8; ++r) {
                const int m   = mBase + waveM * 64 + mt * 16 + r + 8 * g;
                const int bi  = m >> 11;       // m / 2048
                const int seq = m & 2047;
                const float v = acc[mt][nt][r] + bv;
                if (mode == 0) {
                    const int h = nc >> 6, d = nc & 63;
                    outQ[(((size_t)(bi * NH + h)) * SEQ + seq) * HD + d] =
                        (_Float16)(v * 0.125f);     // fold 1/sqrt(HD)
                } else if (nc < DM) {               // K half
                    const int h = nc >> 6, d = nc & 63;
                    outK[(((size_t)(bi * NH + h)) * SEQ + seq) * HD + d] = (_Float16)v;
                } else {                            // V half, stored transposed
                    const int nv = nc - DM;
                    const int h = nv >> 6, d = nv & 63;
                    outV[(((size_t)(bi * NH + h)) * HD + d) * SEQ + seq] = (_Float16)v;
                }
            }
        }
    }
}

// ---------------------------------------------------------------------------
// Flash attention: block = (b,h, 128-query tile), 8 waves; wave owns 16 rows.
// Key loop of 64, double-buffered K/V tiles in LDS (async-DMA staged when
// available): S = Q K^T (8 WMMA), online softmax, O += P V (8 WMMA).
// ---------------------------------------------------------------------------
__global__ __launch_bounds__(256)
void flash_attn_kernel(const _Float16* __restrict__ Qh,
                       const _Float16* __restrict__ Kh,
                       const _Float16* __restrict__ Vt,
                       float* __restrict__ out)
{
    constexpr int LDK = 72;                        // padded stride: conflict-free
    __shared__ _Float16 Klds[2][64 * LDK];         // K tile [key][d]
    __shared__ _Float16 Vlds[2][64 * LDK];         // V^T tile [d][key]
    __shared__ _Float16 Plds[8 * 16 * LDK];        // per-wave P scratch [m][key]

    const int tid  = threadIdx.x;
    const int lane = tid & 31;
    const int wid  = tid >> 5;
    const int g    = lane >> 4;
    const int l16  = lane & 15;

    const int bh    = blockIdx.y;                  // 0..63
    const int b     = bh >> 4;
    const int h     = bh & 15;
    const int qBase = blockIdx.x * 128 + wid * 16;

    // --- per-thread staging coordinates: 2 x 16B chunks each for K and V ---
    const int r0 = tid >> 3,          c0 = tid & 7;          // chunk 0
    const int r1 = (tid + 256) >> 3,  c1 = tid & 7;          // chunk 1 (row + 32)
    const _Float16* kSrc0 = Kh + ((size_t)bh * SEQ + r0) * HD + c0 * 8;
    const _Float16* kSrc1 = Kh + ((size_t)bh * SEQ + r1) * HD + c1 * 8;
    const _Float16* vSrc0 = Vt + ((size_t)bh * HD + r0) * SEQ + c0 * 8;
    const _Float16* vSrc1 = Vt + ((size_t)bh * HD + r1) * SEQ + c1 * 8;
    const int kDstOff0 = r0 * LDK + c0 * 8, kDstOff1 = r1 * LDK + c1 * 8;

    auto stageTile = [&](int buf) {                // issue 4 x 16B copies
        cp16_g2l(kSrc0, &Klds[buf][kDstOff0]);
        cp16_g2l(kSrc1, &Klds[buf][kDstOff1]);
        cp16_g2l(vSrc0, &Vlds[buf][kDstOff0]);
        cp16_g2l(vSrc1, &Vlds[buf][kDstOff1]);
    };

    // Q fragments resident for entire kernel (row = l16, 2 chunks of d=32)
    HFrag qa[2];
    {
        const _Float16* qrow = Qh + ((size_t)bh * SEQ + qBase + l16) * HD;
#pragma unroll
        for (int c = 0; c < 2; ++c) {
            qa[c].p.lo = *(const v8h*)(qrow + c * 32 + g * 8);
            qa[c].p.hi = *(const v8h*)(qrow + c * 32 + 16 + g * 8);
        }
    }

    v8f o[4];
#pragma unroll
    for (int nt = 0; nt < 4; ++nt)
        o[nt] = (v8f){0.f, 0.f, 0.f, 0.f, 0.f, 0.f, 0.f, 0.f};
    float mi[8], li[8];
#pragma unroll
    for (int r = 0; r < 8; ++r) { mi[r] = -1e30f; li[r] = 0.f; }

    _Float16* pl = &Plds[wid * 16 * LDK];

    // prologue: stage key-tile 0 into buffer 0
    stageTile(0);
    cp_fence_lds();
    __syncthreads();

    constexpr int nIter = SEQ / 64;                // 32 key tiles
    for (int it = 0; it < nIter; ++it) {
        const int  cur  = it & 1;
        const bool more = (it + 1) < nIter;
        if (more) {                                // DMA next tile into back buffer
            kSrc0 += 64 * HD; kSrc1 += 64 * HD;    // K: +64 key rows
            vSrc0 += 64;      vSrc1 += 64;         // V^T: +64 key cols
            stageTile(cur ^ 1);
        }

        // ---- S = Q K^T : 4 key tiles x 2 d-chunks ----
        v8f s[4];
#pragma unroll
        for (int nt = 0; nt < 4; ++nt) {
            const int krow = nt * 16 + l16;
            HFrag bk0, bk1;
            bk0.p.lo = *(const v8h*)&Klds[cur][krow * LDK + g * 8];
            bk0.p.hi = *(const v8h*)&Klds[cur][krow * LDK + 16 + g * 8];
            bk1.p.lo = *(const v8h*)&Klds[cur][krow * LDK + 32 + g * 8];
            bk1.p.hi = *(const v8h*)&Klds[cur][krow * LDK + 32 + 16 + g * 8];
            v8f z = (v8f){0.f, 0.f, 0.f, 0.f, 0.f, 0.f, 0.f, 0.f};
            s[nt] = WMMA16(qa[0].v, bk0.v, z);
            s[nt] = WMMA16(qa[1].v, bk1.v, s[nt]);
        }

        // ---- online softmax (row r lives across 16 lanes of a group) ----
        float corr[8];
#pragma unroll
        for (int r = 0; r < 8; ++r) {
            float mx = s[0][r];
#pragma unroll
            for (int nt = 1; nt < 4; ++nt) mx = fmaxf(mx, s[nt][r]);
#pragma unroll
            for (int msk = 1; msk < 16; msk <<= 1)
                mx = fmaxf(mx, __shfl_xor(mx, msk, 32));
            const float mnew = fmaxf(mi[r], mx);
            corr[r] = __expf(mi[r] - mnew);
            mi[r] = mnew;
            float rs = 0.f;
#pragma unroll
            for (int nt = 0; nt < 4; ++nt) {
                const float p = __expf(s[nt][r] - mnew);
                s[nt][r] = p;
                rs += p;
            }
#pragma unroll
            for (int msk = 1; msk < 16; msk <<= 1)
                rs += __shfl_xor(rs, msk, 32);
            li[r] = li[r] * corr[r] + rs;
        }
#pragma unroll
        for (int nt = 0; nt < 4; ++nt)
#pragma unroll
            for (int r = 0; r < 8; ++r)
                o[nt][r] *= corr[r];

        // ---- P: C-layout -> A-layout via per-wave LDS patch ----
#pragma unroll
        for (int nt = 0; nt < 4; ++nt)
#pragma unroll
            for (int r = 0; r < 8; ++r)
                pl[(r + 8 * g) * LDK + nt * 16 + l16] = (_Float16)s[nt][r];
        // same-wave DS ops are in-order: safe to read back without barrier
        HFrag pa[2];
#pragma unroll
        for (int c = 0; c < 2; ++c) {
            pa[c].p.lo = *(const v8h*)&pl[l16 * LDK + c * 32 + g * 8];
            pa[c].p.hi = *(const v8h*)&pl[l16 * LDK + c * 32 + 16 + g * 8];
        }

        // ---- O += P V : 4 d tiles x 2 key-chunks ----
#pragma unroll
        for (int nt = 0; nt < 4; ++nt) {
            const int drow = nt * 16 + l16;
            HFrag bv0, bv1;
            bv0.p.lo = *(const v8h*)&Vlds[cur][drow * LDK + g * 8];
            bv0.p.hi = *(const v8h*)&Vlds[cur][drow * LDK + 16 + g * 8];
            bv1.p.lo = *(const v8h*)&Vlds[cur][drow * LDK + 32 + g * 8];
            bv1.p.hi = *(const v8h*)&Vlds[cur][drow * LDK + 32 + 16 + g * 8];
            o[nt] = WMMA16(pa[0].v, bv0.v, o[nt]);
            o[nt] = WMMA16(pa[1].v, bv1.v, o[nt]);
        }

        if (more) cp_fence_lds();                  // next tile fully in LDS
        __syncthreads();
    }

    // ---- epilogue: normalize and store f32, coalesced 4B stores ----
#pragma unroll
    for (int r = 0; r < 8; ++r) {
        const float inv = 1.0f / li[r];
        const int row = qBase + r + 8 * g;
#pragma unroll
        for (int nt = 0; nt < 4; ++nt) {
            const int col = h * HD + nt * 16 + l16;
            out[((size_t)b * SEQ + row) * DM + col] = o[nt][r] * inv;
        }
    }
}

// ---------------------------------------------------------------------------
extern "C" void kernel_launch(void* const* d_in, const int* in_sizes, int n_in,
                              void* d_out, int out_size, void* d_ws, size_t ws_size,
                              hipStream_t stream)
{
    const float* x    = (const float*)d_in[0];   // [B, LQ, D]
    const float* text = (const float*)d_in[1];   // [B, LK, D]
    const float* Wq   = (const float*)d_in[2];   // [D, D]
    const float* bq   = (const float*)d_in[3];   // [D]
    const float* Wkv  = (const float*)d_in[4];   // [D, 2D]
    const float* bkv  = (const float*)d_in[5];   // [2D]
    float* out        = (float*)d_out;           // [B, LQ, D]

    const size_t headElems = (size_t)NB * NH * SEQ * HD;   // 8M elems, 16MB f16
    _Float16* Qh = (_Float16*)d_ws;
    _Float16* Kh = Qh + headElems;
    _Float16* Vt = Kh + headElems;

    dim3 blk(256);
    // Q = x @ Wq + bq  (M=8192, N=1024, K=1024)
    gemm_proj_kernel<<<dim3(DM / 128, (NB * SEQ) / 128), blk, 0, stream>>>(
        x, Wq, bq, Qh, nullptr, nullptr, DM, DM, 0);
    // KV = text @ Wkv + bkv  (M=8192, N=2048, K=1024)
    gemm_proj_kernel<<<dim3((2 * DM) / 128, (NB * SEQ) / 128), blk, 0, stream>>>(
        text, Wkv, bkv, nullptr, Kh, Vt, 2 * DM, DM, 1);
    // attention
    flash_attn_kernel<<<dim3(SEQ / 128, NB * NH), blk, 0, stream>>>(Qh, Kh, Vt, out);
}
